// ConvNet_18150531793361
// MI455X (gfx1250) — matrix operand order repaired and checked by reading, not compiled, not run on hardware
//
#include <hip/hip_runtime.h>
#include <math.h>
#include <stdint.h>

typedef __attribute__((ext_vector_type(16))) _Float16 v16h;
typedef __attribute__((ext_vector_type(8)))  _Float16 v8h;
typedef __attribute__((ext_vector_type(8)))  float    v8f;
typedef __attribute__((ext_vector_type(4)))  unsigned int v4u;
typedef __attribute__((ext_vector_type(8)))  int          v8i;
typedef __attribute__((ext_vector_type(4)))  int          v4i;

#define BM 64
#define BK 32
#define TWO_PI_F  6.28318530717958647692f
#define QPI_F     0.78539816339744830962f

#if defined(__has_builtin)
#  if __has_builtin(__builtin_amdgcn_tensor_load_to_lds) && __has_builtin(__builtin_amdgcn_s_wait_tensorcnt)
#    define HAVE_TDM 1
#  endif
#endif

__device__ __forceinline__ int imin(int a, int b) { return a < b ? a : b; }
__device__ __forceinline__ int imax(int a, int b) { return a > b ? a : b; }

// ---------------------------------------------------------------- layout utils
__global__ void nchw_to_nhwc(const float* __restrict__ in, float* __restrict__ out,
                             int B, int C, int H, int W) {
  int idx = blockIdx.x * blockDim.x + threadIdx.x;
  int total = B * C * H * W;
  if (idx >= total) return;
  int x = idx % W; int t = idx / W;
  int y = t % H;  t /= H;
  int c = t % C;  int b = t / C;
  out[(((size_t)b * H + y) * W + x) * C + c] = in[idx];
}

// w (O,C,3,3) f32 -> wbufT[N][Kpad] f16 (k = c*9 + tap contiguous per n), zero padded
__global__ void convert_weights(const float* __restrict__ w, _Float16* __restrict__ wbufT,
                                int C, int N, int K, int Kpad) {
  int idx = blockIdx.x * blockDim.x + threadIdx.x;
  int total = Kpad * N;
  if (idx >= total) return;
  int k = idx % Kpad;
  int n = idx / Kpad;
  float v = 0.f;
  if (k < K) {
    int c = k / 9;
    int t = k - c * 9;
    v = w[((size_t)n * C + c) * 9 + t];
  }
  wbufT[(size_t)n * Kpad + k] = (_Float16)v;
}

// --------------------------------------------- fused deform-im2col + WMMA GEMM
// xin: NHWC f32 (B,H,W,C); wbufT: f16 [N][Kpad]; out: NHWC f32 (B,H,W,N), ReLU'd
// Workgroup: 256 threads = 8 waves, 4 (M) x 2 (N); each wave: 16 x WN output.
template<int WN>
__launch_bounds__(256)
__global__ void deform_conv_wmma(const float* __restrict__ xin,
                                 const _Float16* __restrict__ wbufT,
                                 float* __restrict__ out,
                                 int H, int W, int C, int N, int K, int Kpad) {
  constexpr int BN = 2 * WN;
  constexpr int NW = WN / 16;          // accumulators (wmma tiles) per wave
  __shared__ __align__(16) _Float16 As[BM][BK + 8];   // 80B row stride
  __shared__ __align__(16) _Float16 Bs[BN][BK + 8];
  __shared__ float s_wy[BM][9], s_wx[BM][9];
  __shared__ int   s_y0[BM][9], s_x0[BM][9];
  __shared__ int   s_pb[BM];

  const int tid  = threadIdx.x;
  const int HW   = H * W;
  const int baseM = blockIdx.x * BM;
  const int baseN = blockIdx.y * BN;

  // ---- per-pixel, per-tap deform geometry (analytic offsets) ----
  const float cyc = 0.5f * (float)H - 0.5f;
  const float cxc = 0.5f * (float)W - 0.5f;
  for (int e = tid; e < BM * 9; e += 256) {
    int i = e / 9, t = e - i * 9;
    int p = baseM + i;
    int b = p / HW, rem = p - b * HW;
    int y = rem / W, x = rem - y * W;
    if (t == 0) s_pb[i] = b * HW;
    float py, px;
    if (t == 4) {               // center tap: zero offset
      py = (float)y; px = (float)x;
    } else {
      // theta = atan2(x-cx, y-cy) mod 2pi, rounded to 1e-4 (round-half-even)
      float theta = atan2f((float)x - cxc, (float)y - cyc);
      if (theta < 0.f) theta += TWO_PI_F;
      theta = rintf(theta * 10000.f) * 1.0e-4f;
      int m = (t < 4) ? t : (t - 1);
      float ang = theta + QPI_F * (float)m;
      // py = y-1+ky + (cos+ay) collapses to y+cos since ay=1-ky (same for x)
      py = (float)y + cosf(ang);
      px = (float)x + sinf(ang);
    }
    float fy0 = floorf(py), fx0 = floorf(px);
    s_wy[i][t] = py - fy0;
    s_wx[i][t] = px - fx0;
    s_y0[i][t] = (int)fy0;
    s_x0[i][t] = (int)fx0;
  }

  v8f acc[NW];
  #pragma unroll
  for (int j = 0; j < NW; ++j) acc[j] = (v8f){};

  const int wv   = tid >> 5;
  const int lane = tid & 31;
  const int mi = (wv & 3) * 16;       // wave's 16-row strip within the 64-row tile
  const int ni = (wv >> 2) * WN;      // wave's WN-col strip within the BN-col tile

#ifdef HAVE_TDM
  const unsigned ldsB = (unsigned)(uintptr_t)&Bs[0][0]; // low 32 bits = LDS byte offset
#endif

  const int nkb = Kpad / BK;
  for (int kb = 0; kb < nkb; ++kb) {
    __syncthreads();

#ifdef HAVE_TDM
    // ---- B tile via Tensor Data Mover: 2D tile BK x BN from wbufT[N][Kpad],
    //      LDS-padded 16 data DWORDs + 4 pad DWORDs per row == (BK+8) halves ----
    if (wv == 0) {
      uint64_t ga = (uint64_t)(uintptr_t)(wbufT + (size_t)baseN * Kpad + (size_t)kb * BK);
      v4u g0;
      g0[0] = 1u;                                        // count=1, user mode
      g0[1] = ldsB;                                      // lds_addr (bytes)
      g0[2] = (unsigned)(ga & 0xFFFFFFFFu);              // global_addr[31:0]
      g0[3] = (unsigned)((ga >> 32) & 0x01FFFFFFu) | (2u << 30);   // addr[56:32], type=2
      v8i g1;
      g1[0] = (int)((1u << 16)      // data_size = 2 bytes
                  | (1u << 20)      // pad_enable
                  | (3u << 22)      // pad_interval: 16 DWORDs of data
                  | (3u << 25));    // pad_amount: 4 DWORDs of pad
      g1[1] = (int)(((unsigned)Kpad & 0xFFFFu) << 16);                 // tensor_dim0 lo16
      g1[2] = (int)((((unsigned)Kpad >> 16) & 0xFFFFu)                 // tensor_dim0 hi16
                  | (((unsigned)N & 0xFFFFu) << 16));                  // tensor_dim1 lo16
      g1[3] = (int)((((unsigned)N >> 16) & 0xFFFFu)                    // tensor_dim1 hi16
                  | ((unsigned)BK << 16));                             // tile_dim0 = BK
      g1[4] = BN;                                                      // tile_dim1 = BN
      g1[5] = Kpad;                                                    // dim0_stride lo32
      g1[6] = 0;                                                       // stride hi / dim1_stride lo
      g1[7] = 0;
      v4i z4 = (v4i){0, 0, 0, 0};
      v8i z8 = (v8i){0, 0, 0, 0, 0, 0, 0, 0};
      // 6-arg (clang-23 / therock) form: groups 0..3 + aux group + cpol
      __builtin_amdgcn_tensor_load_to_lds(g0, g1, z4, z4, z8, 0);
    }
#endif

    // ---- A tile: 64 pixels x 32 k (bilinear deform samples), f16 ----
    {
      int i   = tid >> 2;
      int kk0 = (tid & 3) * 8;
      int pb  = s_pb[i];
      #pragma unroll
      for (int j = 0; j < 8; ++j) {
        int kk = kk0 + j;
        int k  = kb * BK + kk;
        float v = 0.f;
        if (k < K) {
          int c = k / 9;
          int t = k - c * 9;
          float wy = s_wy[i][t], wx = s_wx[i][t];
          int y0 = s_y0[i][t], x0 = s_x0[i][t];
          int y1 = y0 + 1, x1 = x0 + 1;
          int cy0 = imin(imax(y0, 0), H - 1), cy1 = imin(imax(y1, 0), H - 1);
          int cx0 = imin(imax(x0, 0), W - 1), cx1 = imin(imax(x1, 0), W - 1);
          bool vy0 = (y0 >= 0) && (y0 < H);
          bool vy1 = (y1 >= 0) && (y1 < H);
          bool vx0 = (x0 >= 0) && (x0 < W);
          bool vx1 = (x1 >= 0) && (x1 < W);
          float v00 = (vy0 && vx0) ? xin[((size_t)(pb + cy0 * W + cx0)) * C + c] : 0.f;
          float v01 = (vy0 && vx1) ? xin[((size_t)(pb + cy0 * W + cx1)) * C + c] : 0.f;
          float v10 = (vy1 && vx0) ? xin[((size_t)(pb + cy1 * W + cx0)) * C + c] : 0.f;
          float v11 = (vy1 && vx1) ? xin[((size_t)(pb + cy1 * W + cx1)) * C + c] : 0.f;
          v = (1.f - wy) * ((1.f - wx) * v00 + wx * v01)
            +         wy * ((1.f - wx) * v10 + wx * v11);
        }
        As[i][kk] = (_Float16)v;
      }
    }

#ifdef HAVE_TDM
    if (wv == 0) __builtin_amdgcn_s_wait_tensorcnt(0);
#else
    // ---- fallback B tile copy: contiguous 16B chunks, Bs[n][k] row-major per n ----
    for (int e = tid; e < BN * (BK / 8); e += 256) {
      int n = e >> 2, q = e & 3;
      *(v8h*)&Bs[n][q * 8] =
          *(const v8h*)&wbufT[(size_t)(baseN + n) * Kpad + (size_t)kb * BK + q * 8];
    }
#endif
    __syncthreads();

    // ---- fragments per ISA VGPR layouts, then WMMA ----
    // A 16x32 f16: row = lane%16; halves = K{0..7,16..23} (lane<16) / K{8..15,24..31}
    int r  = mi + (lane & 15);
    int ko = (lane < 16) ? 0 : 8;
    v8h alo = *(const v8h*)&As[r][ko];
    v8h ahi = *(const v8h*)&As[r][ko + 16];
    v16h a = __builtin_shufflevector(alo, ahi, 0,1,2,3,4,5,6,7,8,9,10,11,12,13,14,15);

    // B 32x16 f16: col = lane%16; K{0..7,16..23} (lane<16) / K{8..15,24..31}
    int kb2 = (lane < 16) ? 0 : 8;
    #pragma unroll
    for (int j = 0; j < NW; ++j) {
      int n0 = ni + j * 16 + (lane & 15);
      v8h blo = *(const v8h*)&Bs[n0][kb2];
      v8h bhi = *(const v8h*)&Bs[n0][kb2 + 16];
      v16h bf = __builtin_shufflevector(blo, bhi, 0,1,2,3,4,5,6,7,8,9,10,11,12,13,14,15);
      acc[j] = __builtin_amdgcn_wmma_f32_16x16x32_f16(false, a, false, bf,
                                                      (short)0, acc[j], false, false);
    }
  }

  // ---- epilogue: D layout VGPR r -> row r (lanes<16) / 8+r (lanes>=16) ----
  int rowOff = mi + ((lane < 16) ? 0 : 8);
  #pragma unroll
  for (int j = 0; j < NW; ++j) {
    int col = baseN + ni + j * 16 + (lane & 15);
    #pragma unroll
    for (int r2 = 0; r2 < 8; ++r2) {
      size_t m = (size_t)(baseM + rowOff + r2);
      float v = acc[j][r2];
      v = v > 0.f ? v : 0.f;
      out[m * N + col] = v;
    }
  }
}

// ------------------------------------------------------------------- pool / fc
__global__ void maxpool_nhwc(const float* __restrict__ in, float* __restrict__ out,
                             int B, int H, int W, int C) {
  int Ho = H >> 1, Wo = W >> 1;
  int idx = blockIdx.x * blockDim.x + threadIdx.x;
  int total = B * Ho * Wo * C;
  if (idx >= total) return;
  int c = idx % C; int t = idx / C;
  int x = t % Wo; t /= Wo;
  int y = t % Ho; int b = t / Ho;
  size_t base = (((size_t)b * H + 2 * y) * W + 2 * x) * C + c;
  size_t rowS = (size_t)W * C;
  float m0 = in[base],        m1 = in[base + C];
  float m2 = in[base + rowS], m3 = in[base + rowS + C];
  float m = m0 > m1 ? m0 : m1;
  float n = m2 > m3 ? m2 : m3;
  out[idx] = m > n ? m : n;
}

__global__ void avgpool4x4(const float* __restrict__ in, float* __restrict__ out, int B, int C) {
  int idx = blockIdx.x * blockDim.x + threadIdx.x;
  if (idx >= B * C) return;
  int c = idx % C, b = idx / C;
  float s = 0.f;
  for (int p = 0; p < 16; ++p) s += in[((size_t)b * 16 + p) * C + c];
  out[idx] = s * (1.f / 16.f);
}

__global__ void fc_kernel(const float* __restrict__ in, const float* __restrict__ w,
                          const float* __restrict__ bias, float* __restrict__ out,
                          int M, int N, int K, int relu) {
  int idx = blockIdx.x * blockDim.x + threadIdx.x;
  if (idx >= M * N) return;
  int n = idx % N, m = idx / N;
  float s = bias[n];
  const float* ip = in + (size_t)m * K;
  const float* wp = w + (size_t)n * K;
  for (int k = 0; k < K; ++k) s += ip[k] * wp[k];
  if (relu) s = s > 0.f ? s : 0.f;
  out[idx] = s;
}

// --------------------------------------------------------------------- launch
extern "C" void kernel_launch(void* const* d_in, const int* in_sizes, int n_in,
                              void* d_out, int out_size, void* d_ws, size_t ws_size,
                              hipStream_t stream) {
  (void)in_sizes; (void)n_in; (void)out_size; (void)ws_size;
  const float* x = (const float*)d_in[0];
  const float* wconv[13];
  for (int i = 0; i < 13; ++i) wconv[i] = (const float*)d_in[1 + i];
  const float* fc1_w = (const float*)d_in[14];
  const float* fc1_b = (const float*)d_in[15];
  const float* fc2_w = (const float*)d_in[16];
  const float* fc2_b = (const float*)d_in[17];
  const float* fc3_w = (const float*)d_in[18];
  const float* fc3_b = (const float*)d_in[19];
  float* outp = (float*)d_out;

  const int B = 8;
  char* ws = (char*)d_ws;
  const size_t ACT = (size_t)8 * 128 * 128 * 64 * sizeof(float);   // 32 MB
  float*    bufA = (float*)ws;
  float*    bufB = (float*)(ws + ACT);
  _Float16* wbuf = (_Float16*)(ws + 2 * ACT);                      // <= 4.7 MB used
  float*    pooled = (float*)(ws + 2 * ACT + (size_t)8 * 1024 * 1024);
  float*    f1 = pooled + 8 * 512;
  float*    f2 = f1 + 8 * 4096;

  auto conv = [&](const float* in, float* out, const float* w, int H, int W, int C, int N) {
    int K = C * 9;
    int Kpad = (K + BK - 1) & ~(BK - 1);
    int tw = Kpad * N;
    convert_weights<<<(tw + 255) / 256, 256, 0, stream>>>(w, wbuf, C, N, K, Kpad);
    int M = B * H * W;
    if (N >= 128) {   // wide N tile: BN=128, 4 WMMA/wave/step, halves gather redundancy
      dim3 grid(M / BM, N / 128);
      deform_conv_wmma<64><<<grid, 256, 0, stream>>>(in, wbuf, out, H, W, C, N, K, Kpad);
    } else {          // BN=64
      dim3 grid(M / BM, N / 64);
      deform_conv_wmma<32><<<grid, 256, 0, stream>>>(in, wbuf, out, H, W, C, N, K, Kpad);
    }
  };
  auto pool = [&](const float* in, float* out, int H, int W, int C) {
    int total = B * (H / 2) * (W / 2) * C;
    maxpool_nhwc<<<(total + 255) / 256, 256, 0, stream>>>(in, out, B, H, W, C);
  };

  // input NCHW (8,3,128,128) -> NHWC
  {
    int total = B * 3 * 128 * 128;
    nchw_to_nhwc<<<(total + 255) / 256, 256, 0, stream>>>(x, bufA, B, 3, 128, 128);
  }
  // block 1 @128
  conv(bufA, bufB, wconv[0], 128, 128, 3, 64);
  conv(bufB, bufA, wconv[1], 128, 128, 64, 64);
  pool(bufA, bufB, 128, 128, 64);
  // block 2 @64
  conv(bufB, bufA, wconv[2], 64, 64, 64, 128);
  conv(bufA, bufB, wconv[3], 64, 64, 128, 128);
  pool(bufB, bufA, 64, 64, 128);
  // block 3 @32
  conv(bufA, bufB, wconv[4], 32, 32, 128, 256);
  conv(bufB, bufA, wconv[5], 32, 32, 256, 256);
  conv(bufA, bufB, wconv[6], 32, 32, 256, 256);
  pool(bufB, bufA, 32, 32, 256);
  // block 4 @16
  conv(bufA, bufB, wconv[7], 16, 16, 256, 512);
  conv(bufB, bufA, wconv[8], 16, 16, 512, 512);
  conv(bufA, bufB, wconv[9], 16, 16, 512, 512);
  pool(bufB, bufA, 16, 16, 512);
  // block 5 @8
  conv(bufA, bufB, wconv[10], 8, 8, 512, 512);
  conv(bufB, bufA, wconv[11], 8, 8, 512, 512);
  conv(bufA, bufB, wconv[12], 8, 8, 512, 512);
  pool(bufB, bufA, 8, 8, 512);           // -> (8,4,4,512) NHWC in bufA

  avgpool4x4<<<(B * 512 + 255) / 256, 256, 0, stream>>>(bufA, pooled, B, 512);
  fc_kernel<<<(B * 4096 + 255) / 256, 256, 0, stream>>>(pooled, fc1_w, fc1_b, f1, B, 4096, 512, 1);
  fc_kernel<<<(B * 4096 + 255) / 256, 256, 0, stream>>>(f1, fc2_w, fc2_b, f2, B, 4096, 4096, 1);
  fc_kernel<<<(B * 30 + 255) / 256, 256, 0, stream>>>(f2, fc3_w, fc3_b, outp, B, 30, 4096, 0);
}
